// GCNEncoder_68255620268445
// MI455X (gfx1250) — compile-verified, matrix operand order checked by reading, not run on hardware
//
#include <hip/hip_runtime.h>
#include <stdint.h>

#define FDIM   256
#define INDIM  512
#define LN_EPS 1e-5f

typedef __attribute__((ext_vector_type(16))) __bf16 v16bf;
typedef __attribute__((ext_vector_type(8)))  float  v8f;

// ---------- helpers ----------
__device__ __forceinline__ unsigned short f2bf_bits(float f) {
    union { float f; unsigned u; } a; a.f = f;
    unsigned r = a.u + 0x7FFFu + ((a.u >> 16) & 1u);   // round-to-nearest-even
    return (unsigned short)(r >> 16);
}

__device__ __forceinline__ v16bf ldfrag(const __bf16* p, int hofs) {
    // 16-bit WMMA fragment layout (ISA 7.12.2): lane half h needs
    // k = {h*8 .. h*8+7} in VGPRs 0-3 and {16+h*8 .. +7} in VGPRs 4-7,
    // i.e. two contiguous 16-byte chunks.
    union { int4 q[2]; v16bf v; } u;
    u.q[0] = *reinterpret_cast<const int4*>(p + hofs);
    u.q[1] = *reinterpret_cast<const int4*>(p + 16 + hofs);
    return u.v;
}

// ---------- degree / norm ----------
__global__ void deg_init(float* __restrict__ deg, int n) {
    int i = blockIdx.x * blockDim.x + threadIdx.x;
    if (i < n) deg[i] = 1.0f;                 // self-loop
}

__global__ void deg_count(const int* __restrict__ dst, float* __restrict__ deg, int e) {
    int i = blockIdx.x * blockDim.x + threadIdx.x;
    if (i < e) atomicAdd(&deg[dst[i]], 1.0f);
}

__global__ void deg_fin(const float* __restrict__ deg, float* __restrict__ dis,
                        float* __restrict__ invd, int n) {
    int i = blockIdx.x * blockDim.x + threadIdx.x;
    if (i < n) {
        float d = deg[i];
        dis[i]  = rsqrtf(d);
        invd[i] = 1.0f / d;
    }
}

// ---------- fp32 -> bf16 (4 elements / thread) ----------
__global__ void cvt_bf16_4(const float* __restrict__ in, __bf16* __restrict__ out, long long n4) {
    long long i = (long long)blockIdx.x * blockDim.x + threadIdx.x;
    if (i >= n4) return;
    float4 v = *reinterpret_cast<const float4*>(in + i * 4);
    union { unsigned u[2]; uint2 q; } pk;
    pk.u[0] = f2bf_bits(v.x) | ((unsigned)f2bf_bits(v.y) << 16);
    pk.u[1] = f2bf_bits(v.z) | ((unsigned)f2bf_bits(v.w) << 16);
    *reinterpret_cast<uint2*>(out + i * 4) = pk.q;
}

// ---------- W (K x Nout, row-major f32) -> Wt (Nout x K, row-major bf16) ----------
__global__ void prep_w(const float* __restrict__ W, __bf16* __restrict__ Wt, int K, int Nout) {
    int idx = blockIdx.x * blockDim.x + threadIdx.x;
    if (idx >= K * Nout) return;
    int n = idx % Nout, k = idx / Nout;
    unsigned short b = f2bf_bits(W[(size_t)k * Nout + n]);
    *reinterpret_cast<unsigned short*>(Wt + (size_t)n * K + k) = b;
}

// ---------- WMMA GEMM: H[M x Nout] = A[M x K] * Wt[Nout x K]^T ----------
template <int K>
__global__ void __launch_bounds__(256)
gemm_bf16_wmma(const __bf16* __restrict__ A, const __bf16* __restrict__ Wt,
               float* __restrict__ Hout, int M, int Nout) {
    const int lane = threadIdx.x & 31;
    const int wave = threadIdx.x >> 5;
    const int nColT = Nout >> 6;                       // 64-wide column tiles
    const int wid = blockIdx.x * 8 + wave;
    const int mT = wid / nColT;
    const int nT = wid - mT * nColT;
    const int row0 = mT << 4;
    if (row0 >= M) return;
    const int col0 = nT << 6;
    const int m = lane & 15;
    const int hofs = (lane >> 4) * 8;

    const __bf16* aP  = A  + (size_t)(row0 + m) * K;
    const __bf16* bP0 = Wt + (size_t)(col0 +  0 + m) * K;
    const __bf16* bP1 = Wt + (size_t)(col0 + 16 + m) * K;
    const __bf16* bP2 = Wt + (size_t)(col0 + 32 + m) * K;
    const __bf16* bP3 = Wt + (size_t)(col0 + 48 + m) * K;

    v8f acc0 = {}, acc1 = {}, acc2 = {}, acc3 = {};

#pragma unroll
    for (int k0 = 0; k0 < K; k0 += 32) {
        __builtin_prefetch(aP + k0 + 128, 0, 0);       // global_prefetch_b8, speculative
        v16bf a  = ldfrag(aP  + k0, hofs);
        v16bf b0 = ldfrag(bP0 + k0, hofs);
        v16bf b1 = ldfrag(bP1 + k0, hofs);
        v16bf b2 = ldfrag(bP2 + k0, hofs);
        v16bf b3 = ldfrag(bP3 + k0, hofs);
        acc0 = __builtin_amdgcn_wmma_f32_16x16x32_bf16(false, a, false, b0, (short)0, acc0, false, false);
        acc1 = __builtin_amdgcn_wmma_f32_16x16x32_bf16(false, a, false, b1, (short)0, acc1, false, false);
        acc2 = __builtin_amdgcn_wmma_f32_16x16x32_bf16(false, a, false, b2, (short)0, acc2, false, false);
        acc3 = __builtin_amdgcn_wmma_f32_16x16x32_bf16(false, a, false, b3, (short)0, acc3, false, false);
    }

    // D layout: VGPR r -> row (hofs + r), N = lane&15
    float* o = Hout + (size_t)(row0 + hofs) * Nout + col0 + m;
#pragma unroll
    for (int r = 0; r < 8; ++r) {
        o[(size_t)r * Nout +  0] = acc0[r];
        o[(size_t)r * Nout + 16] = acc1[r];
        o[(size_t)r * Nout + 32] = acc2[r];
        o[(size_t)r * Nout + 48] = acc3[r];
    }
}

// ---------- edge scatter: AGG[dst] += dis[src]*dis[dst] * H[src] ----------
__global__ void edge_scatter(const int* __restrict__ src, const int* __restrict__ dst,
                             const float* __restrict__ dis, const float* __restrict__ H,
                             float* __restrict__ AGG, int e) {
    long long gid = (long long)blockIdx.x * blockDim.x + threadIdx.x;
    if (gid >= (long long)e * 64) return;
    int ei = (int)(gid >> 6);
    int j  = ((int)gid & 63) * 4;
    int s = src[ei], d = dst[ei];
    float w = dis[s] * dis[d];
    float4 hv = *reinterpret_cast<const float4*>(H + (size_t)s * FDIM + j);
    float* out = AGG + (size_t)d * FDIM + j;
    atomicAdd(out + 0, w * hv.x);
    atomicAdd(out + 1, w * hv.y);
    atomicAdd(out + 2, w * hv.z);
    atomicAdd(out + 3, w * hv.w);
}

// ---------- combine: agg + h*inv_deg + b, ReLU, optional LayerNorm; emit f32 and/or bf16 ----------
__global__ void __launch_bounds__(256)
combine_kernel(const float* __restrict__ AGG, const float* __restrict__ H,
               const float* __restrict__ invd, const float* __restrict__ bias,
               const float* __restrict__ gamma, const float* __restrict__ beta,
               float* __restrict__ outF, __bf16* __restrict__ outB, int n, int doLN) {
    const int lane = threadIdx.x & 31;
    const int wave = threadIdx.x >> 5;
    const int row  = blockIdx.x * 8 + wave;
    if (row >= n) return;

    const float id = invd[row];
    const size_t base = (size_t)row * FDIM + lane * 8;
    float v[8];
#pragma unroll
    for (int t = 0; t < 2; ++t) {
        float4 a  = *reinterpret_cast<const float4*>(AGG  + base + t * 4);
        float4 hh = *reinterpret_cast<const float4*>(H    + base + t * 4);
        float4 bb = *reinterpret_cast<const float4*>(bias + lane * 8 + t * 4);
        v[t*4+0] = fmaxf(a.x + hh.x * id + bb.x, 0.0f);
        v[t*4+1] = fmaxf(a.y + hh.y * id + bb.y, 0.0f);
        v[t*4+2] = fmaxf(a.z + hh.z * id + bb.z, 0.0f);
        v[t*4+3] = fmaxf(a.w + hh.w * id + bb.w, 0.0f);
    }

    if (doLN) {
        float s = 0.0f, ss = 0.0f;
#pragma unroll
        for (int i = 0; i < 8; ++i) { s += v[i]; ss += v[i] * v[i]; }
#pragma unroll
        for (int off = 16; off > 0; off >>= 1) {
            s  += __shfl_xor(s,  off, 32);
            ss += __shfl_xor(ss, off, 32);
        }
        const float mu  = s * (1.0f / FDIM);
        const float var = ss * (1.0f / FDIM) - mu * mu;
        const float rs  = rsqrtf(var + LN_EPS);
#pragma unroll
        for (int t = 0; t < 2; ++t) {
            float4 g = *reinterpret_cast<const float4*>(gamma + lane * 8 + t * 4);
            float4 b = *reinterpret_cast<const float4*>(beta  + lane * 8 + t * 4);
            v[t*4+0] = (v[t*4+0] - mu) * rs * g.x + b.x;
            v[t*4+1] = (v[t*4+1] - mu) * rs * g.y + b.y;
            v[t*4+2] = (v[t*4+2] - mu) * rs * g.z + b.z;
            v[t*4+3] = (v[t*4+3] - mu) * rs * g.w + b.w;
        }
    }

    if (outF) {
#pragma unroll
        for (int t = 0; t < 2; ++t) {
            float4 o; o.x = v[t*4+0]; o.y = v[t*4+1]; o.z = v[t*4+2]; o.w = v[t*4+3];
            *reinterpret_cast<float4*>(outF + base + t * 4) = o;
        }
    }
    if (outB) {
        union { unsigned u[4]; int4 q; } pk;
#pragma unroll
        for (int t = 0; t < 4; ++t)
            pk.u[t] = f2bf_bits(v[2*t]) | ((unsigned)f2bf_bits(v[2*t+1]) << 16);
        *reinterpret_cast<int4*>(outB + base) = pk.q;
    }
}

// ---------- host launcher ----------
extern "C" void kernel_launch(void* const* d_in, const int* in_sizes, int n_in,
                              void* d_out, int out_size, void* d_ws, size_t ws_size,
                              hipStream_t stream) {
    const float* x   = (const float*)d_in[0];
    const int*   ei  = (const int*)d_in[1];
    const float* W1  = (const float*)d_in[2];
    const float* b1  = (const float*)d_in[3];
    const float* W2  = (const float*)d_in[4];
    const float* b2  = (const float*)d_in[5];
    const float* W3  = (const float*)d_in[6];
    const float* b3  = (const float*)d_in[7];
    const float* g1  = (const float*)d_in[8];
    const float* be1 = (const float*)d_in[9];
    const float* g2  = (const float*)d_in[10];
    const float* be2 = (const float*)d_in[11];

    const int N = in_sizes[0] / INDIM;   // 100000
    const int E = in_sizes[1] / 2;       // 1600000
    const int* srcIdx = ei;
    const int* dstIdx = ei + E;

    char* ws = (char*)d_ws;
    size_t cur = 0;
    auto walloc = [&](size_t bytes) -> char* {
        char* p = ws + cur;
        cur += (bytes + 255) & ~(size_t)255;
        return p;
    };
    float*  deg  = (float*) walloc((size_t)N * 4);
    float*  dis  = (float*) walloc((size_t)N * 4);
    float*  invd = (float*) walloc((size_t)N * 4);
    __bf16* Xb   = (__bf16*)walloc((size_t)N * INDIM * 2);
    __bf16* Hb   = (__bf16*)walloc((size_t)N * FDIM * 2);
    float*  Hd   = (float*) walloc((size_t)N * FDIM * 4);
    float*  AGG  = (float*) walloc((size_t)N * FDIM * 4);
    __bf16* Wt   = (__bf16*)walloc((size_t)INDIM * FDIM * 2);
    (void)ws_size; (void)n_in; (void)out_size;

    const dim3 blk(256);

    deg_init <<<(N + 255) / 256, blk, 0, stream>>>(deg, N);
    deg_count<<<(E + 255) / 256, blk, 0, stream>>>(dstIdx, deg, E);
    deg_fin  <<<(N + 255) / 256, blk, 0, stream>>>(deg, dis, invd, N);

    {
        long long n4 = (long long)N * INDIM / 4;
        cvt_bf16_4<<<(unsigned)((n4 + 255) / 256), blk, 0, stream>>>(x, Xb, n4);
    }

    const int wavesGemm = (N / 16) * (FDIM / 64);
    const unsigned gemmBlocks = (unsigned)((wavesGemm + 7) / 8);
    const long long edgeThreads = (long long)E * 64;
    const unsigned edgeBlocks = (unsigned)((edgeThreads + 255) / 256);
    const unsigned combBlocks = (unsigned)((N + 7) / 8);

    // ---- layer 1 (K = 512) ----
    prep_w<<<(INDIM * FDIM + 255) / 256, blk, 0, stream>>>(W1, Wt, INDIM, FDIM);
    gemm_bf16_wmma<INDIM><<<gemmBlocks, blk, 0, stream>>>(Xb, Wt, Hd, N, FDIM);
    hipMemsetAsync(AGG, 0, (size_t)N * FDIM * 4, stream);
    edge_scatter<<<edgeBlocks, blk, 0, stream>>>(srcIdx, dstIdx, dis, Hd, AGG, E);
    combine_kernel<<<combBlocks, blk, 0, stream>>>(AGG, Hd, invd, b1, g1, be1, nullptr, Hb, N, 1);

    // ---- layer 2 (K = 256) ----
    prep_w<<<(FDIM * FDIM + 255) / 256, blk, 0, stream>>>(W2, Wt, FDIM, FDIM);
    gemm_bf16_wmma<FDIM><<<gemmBlocks, blk, 0, stream>>>(Hb, Wt, Hd, N, FDIM);
    hipMemsetAsync(AGG, 0, (size_t)N * FDIM * 4, stream);
    edge_scatter<<<edgeBlocks, blk, 0, stream>>>(srcIdx, dstIdx, dis, Hd, AGG, E);
    combine_kernel<<<combBlocks, blk, 0, stream>>>(AGG, Hd, invd, b2, g2, be2, nullptr, Hb, N, 1);

    // ---- layer 3 (K = 256, ReLU only, fp32 out) ----
    prep_w<<<(FDIM * FDIM + 255) / 256, blk, 0, stream>>>(W3, Wt, FDIM, FDIM);
    gemm_bf16_wmma<FDIM><<<gemmBlocks, blk, 0, stream>>>(Hb, Wt, Hd, N, FDIM);
    hipMemsetAsync(AGG, 0, (size_t)N * FDIM * 4, stream);
    edge_scatter<<<edgeBlocks, blk, 0, stream>>>(srcIdx, dstIdx, dis, Hd, AGG, E);
    combine_kernel<<<combBlocks, blk, 0, stream>>>(AGG, Hd, invd, b3, nullptr, nullptr,
                                                   (float*)d_out, nullptr, N, 0);
}
